// GRU_74328704025086
// MI455X (gfx1250) — matrix operand order repaired and checked by reading, not compiled, not run on hardware
//
#include <hip/hip_runtime.h>

#define T_STEPS 512
#define BATCH   64
#define IN      512
#define HID     1024
#define KTOT    (HID + IN)   // 1536

typedef __attribute__((ext_vector_type(16))) __bf16        v16bf;
typedef __attribute__((ext_vector_type(8)))  float         v8f;
typedef __attribute__((ext_vector_type(4)))  unsigned int  v4u;
typedef __attribute__((ext_vector_type(4)))  float         v4f;

union ABfrag {
    v16bf          v;
    v4u            q[2];
    unsigned short u[16];
};

__device__ __forceinline__ unsigned short f32_to_bf16(float f) {
    unsigned int u = __float_as_uint(f);
    u += 0x7FFFu + ((u >> 16) & 1u);          // round-to-nearest-even
    return (unsigned short)(u >> 16);
}

// A fragment from a bf16 row-major buffer with row stride `stride`.
// lane<16: K = c1..c1+7 and c1+16..c1+23 (c1 = k0);  lane>=16: c1 = k0+8.
__device__ __forceinline__ v16bf load_a_bf16(const unsigned short* __restrict__ A,
                                             int row, int stride, int c1) {
    ABfrag f;
    const unsigned short* p = A + row * stride + c1;
    f.q[0] = *(const v4u*)(p);
    f.q[1] = *(const v4u*)(p + 16);
    return f.v;
}

// B fragment: B[k][n] = W[n][k], W bf16 row-major [HID][KTOT].
// lane holds column n = n0 + (lane&15), K = koff..koff+7 and koff+16..koff+23 (koff = k0 + 8*(lane>>4)).
__device__ __forceinline__ v16bf load_b_w(const unsigned short* __restrict__ W,
                                          int wrow, int koff) {
    ABfrag f;
    const unsigned short* p = W + wrow * KTOT + koff;
    f.q[0] = *(const v4u*)(p);
    f.q[1] = *(const v4u*)(p + 16);
    return f.v;
}

// ---- step kernel 1: g = sigmoid([h, x_t] @ Wz^T + bz); Z=R=g; rh = g*h ----
__global__ __launch_bounds__(128) void gru_step_z(
    const unsigned short* __restrict__ xb_t, const unsigned short* __restrict__ Wzb,
    const float* __restrict__ bz, const float* __restrict__ Hprev,
    const unsigned short* __restrict__ hb, float* __restrict__ Zout,
    float* __restrict__ Rout, unsigned short* __restrict__ rhb) {
    const int lane = threadIdx.x & 31;
    const int wave = threadIdx.x >> 5;
    const int tile = blockIdx.x * 4 + wave;     // 256 tiles: 4 (M) x 64 (N)
    const int m0 = (tile & 3) * 16;
    const int n0 = (tile >> 2) * 16;
    const int lh = lane & 15;
    const int hi = lane >> 4;
    const int arow = m0 + lh;
    const int wrow = n0 + lh;

    v8f acc = {};
#pragma unroll 4
    for (int k0 = 0; k0 < HID; k0 += 32) {      // h-region of concat
        v16bf a = load_a_bf16(hb, arow, HID, k0 + hi * 8);
        v16bf b = load_b_w(Wzb, wrow, k0 + hi * 8);
        acc = __builtin_amdgcn_wmma_f32_16x16x32_bf16(false, a, false, b,
                                                      (short)0, acc, false, false);
    }
#pragma unroll 4
    for (int k0 = 0; k0 < IN; k0 += 32) {       // x-region of concat (pre-converted bf16)
        v16bf a = load_a_bf16(xb_t, arow, IN, k0 + hi * 8);
        v16bf b = load_b_w(Wzb, wrow, HID + k0 + hi * 8);
        acc = __builtin_amdgcn_wmma_f32_16x16x32_bf16(false, a, false, b,
                                                      (short)0, acc, false, false);
    }

    const int col = n0 + lh;
    const float bzv = bz[col];
#pragma unroll
    for (int i = 0; i < 8; ++i) {
        const int row = m0 + hi * 8 + i;
        const int idx = row * HID + col;
        float g = 1.0f / (1.0f + __expf(-(acc[i] + bzv)));
        Zout[idx] = g;
        Rout[idx] = g;
        rhb[idx]  = f32_to_bf16(g * Hprev[idx]);
    }
}

// ---- step kernel 2: cand = tanh([rh, x_t] @ Wh^T + bh); h' = (1-g)h + g*cand ----
__global__ __launch_bounds__(128) void gru_step_h(
    const unsigned short* __restrict__ xb_t, const unsigned short* __restrict__ Whb,
    const float* __restrict__ bh, const float* __restrict__ Hprev,
    const float* __restrict__ Zt, const unsigned short* __restrict__ rhb,
    float* __restrict__ Hnext, unsigned short* __restrict__ hb) {
    const int lane = threadIdx.x & 31;
    const int wave = threadIdx.x >> 5;
    const int tile = blockIdx.x * 4 + wave;
    const int m0 = (tile & 3) * 16;
    const int n0 = (tile >> 2) * 16;
    const int lh = lane & 15;
    const int hi = lane >> 4;
    const int arow = m0 + lh;
    const int wrow = n0 + lh;

    v8f acc = {};
#pragma unroll 4
    for (int k0 = 0; k0 < HID; k0 += 32) {      // r*h region
        v16bf a = load_a_bf16(rhb, arow, HID, k0 + hi * 8);
        v16bf b = load_b_w(Whb, wrow, k0 + hi * 8);
        acc = __builtin_amdgcn_wmma_f32_16x16x32_bf16(false, a, false, b,
                                                      (short)0, acc, false, false);
    }
#pragma unroll 4
    for (int k0 = 0; k0 < IN; k0 += 32) {       // x region (pre-converted bf16)
        v16bf a = load_a_bf16(xb_t, arow, IN, k0 + hi * 8);
        v16bf b = load_b_w(Whb, wrow, HID + k0 + hi * 8);
        acc = __builtin_amdgcn_wmma_f32_16x16x32_bf16(false, a, false, b,
                                                      (short)0, acc, false, false);
    }

    const int col = n0 + lh;
    const float bhv = bh[col];
#pragma unroll
    for (int i = 0; i < 8; ++i) {
        const int row = m0 + hi * 8 + i;
        const int idx = row * HID + col;
        float cand = tanhf(acc[i] + bhv);
        float g  = Zt[idx];
        float hn = (1.0f - g) * Hprev[idx] + g * cand;
        Hnext[idx] = hn;
        hb[idx]    = f32_to_bf16(hn);
    }
}

// One-time conversion: weights f32 -> bf16 (elementwise).
__global__ void gru_cvt_weights(const float* __restrict__ Wz,
                                const float* __restrict__ Wh,
                                unsigned short* __restrict__ wzb,
                                unsigned short* __restrict__ whb, int n) {
    int i = blockIdx.x * blockDim.x + threadIdx.x;
    if (i < n) {
        wzb[i] = f32_to_bf16(Wz[i]);
        whb[i] = f32_to_bf16(Wh[i]);
    }
}

// One-time conversion: x f32 -> bf16, 8 elements per thread (b128 in, b128 out).
__global__ void gru_cvt_x(const float* __restrict__ X,
                          unsigned short* __restrict__ Xb, int n8) {
    int i = blockIdx.x * blockDim.x + threadIdx.x;
    if (i < n8) {
        const float* p = X + (size_t)i * 8;
        v4f a = *(const v4f*)(p);
        v4f b = *(const v4f*)(p + 4);
        ABfrag f;
#pragma unroll
        for (int j = 0; j < 4; ++j) {
            f.u[j]     = f32_to_bf16(a[j]);
            f.u[4 + j] = f32_to_bf16(b[j]);
        }
        *(v4u*)(Xb + (size_t)i * 8) = f.q[0];
    }
}

__global__ void gru_init(float* __restrict__ H0, unsigned short* __restrict__ hb, int n) {
    int i = blockIdx.x * blockDim.x + threadIdx.x;
    if (i < n) { H0[i] = 0.0f; hb[i] = 0; }
}

extern "C" void kernel_launch(void* const* d_in, const int* in_sizes, int n_in,
                              void* d_out, int out_size, void* d_ws, size_t ws_size,
                              hipStream_t stream) {
    const float* x  = (const float*)d_in[0];
    const float* Wz = (const float*)d_in[1];
    const float* bz = (const float*)d_in[2];
    const float* Wh = (const float*)d_in[3];
    const float* bh = (const float*)d_in[4];

    const size_t SBH = (size_t)BATCH * HID;         // 65536 elements / step
    float* H = (float*)d_out;                       // [(T+1), B, HID]
    float* Z = H + (size_t)(T_STEPS + 1) * SBH;     // [T, B, HID]
    float* R = Z + (size_t)T_STEPS * SBH;           // [T, B, HID]

    unsigned short* wzb = (unsigned short*)d_ws;            // 3 MB
    unsigned short* whb = wzb + (size_t)HID * KTOT;         // 3 MB
    unsigned short* hb  = whb + (size_t)HID * KTOT;         // 128 KB
    unsigned short* rhb = hb + SBH;                         // 128 KB
    unsigned short* xb  = rhb + SBH;                        // 32 MB: x in bf16

    const int wn = HID * KTOT;
    gru_cvt_weights<<<(wn + 255) / 256, 256, 0, stream>>>(Wz, Wh, wzb, whb, wn);
    const int xn8 = (T_STEPS * BATCH * IN) / 8;
    gru_cvt_x<<<(xn8 + 255) / 256, 256, 0, stream>>>(x, xb, xn8);
    gru_init<<<((int)SBH + 255) / 256, 256, 0, stream>>>(H, hb, (int)SBH);

    for (int t = 0; t < T_STEPS; ++t) {
        const unsigned short* xb_t = xb + (size_t)t * BATCH * IN;
        float* Hp = H + (size_t)t * SBH;
        float* Hn = Hp + SBH;
        float* Zt = Z + (size_t)t * SBH;
        float* Rt = R + (size_t)t * SBH;
        gru_step_z<<<64, 128, 0, stream>>>(xb_t, wzb, bz, Hp, hb, Zt, Rt, rhb);
        gru_step_h<<<64, 128, 0, stream>>>(xb_t, whb, bh, Hp, Zt, rhb, Hn, hb);
    }
}
